// PointMatching_56831007261100
// MI455X (gfx1250) — compile-verified, matrix operand order checked by reading, not run on hardware
//
#include <hip/hip_runtime.h>
#include <hip/hip_bf16.h>

// ---------------- problem constants (match reference) ----------------
#define BB   2
#define NF   20000
#define DD   64
#define NCC  1024
#define NC   256
#define PP   64
#define KTOP 3
#define RAD  0.15f
#define THC  0.05f
#define ITERS 100
#define NEGV (-1e9f)
#define AUGN 65        // P+1
#define CAND_CAP 512

typedef __attribute__((ext_vector_type(16))) _Float16 v16h;
typedef __attribute__((ext_vector_type(8)))  _Float16 v8h;
typedef __attribute__((ext_vector_type(8)))  float    v8f;

// =====================================================================
// Kernel 1: patch gather.  One block per (b,c). Histogram radix-select
// of the P=64 closest in-radius points among NF=20000.
// =====================================================================
__global__ __launch_bounds__(256)
void gather_kernel(const float* __restrict__ feats,   // (B,NF,D)
                   const float* __restrict__ points,  // (B,NF,3)
                   const int*   __restrict__ spi,     // (B,NC)
                   const float* __restrict__ coarse,  // (B,NCC,3)
                   float* __restrict__ ftOut,         // (B,NC,P,D)
                   float* __restrict__ ptOut,         // (B,NC,P,3)
                   int*   __restrict__ maskOut)       // (B,NC,P)
{
    __shared__ unsigned int hist[4096];
    __shared__ unsigned int scanp[256];
    __shared__ unsigned int candKey[CAND_CAP];
    __shared__ unsigned int candIdx[CAND_CAP];
    __shared__ unsigned int candCnt;
    __shared__ unsigned int s_T;
    __shared__ int selIdx[PP];
    __shared__ float cx, cy, cz;

    const int bc  = blockIdx.x;          // 0..B*NC-1
    const int b   = bc / NC;
    const int tid = threadIdx.x;

    // init
    for (int i = tid; i < 4096; i += 256) hist[i] = 0u;
    if (tid < PP) selIdx[tid] = -1;
    if (tid == 0) {
        candCnt = 0u;
        s_T = 4095u;
        int ci = spi[bc];
        const float* cp = coarse + ((size_t)b * NCC + ci) * 3;
        cx = cp[0]; cy = cp[1]; cz = cp[2];
    }
    __syncthreads();

    const float r2 = RAD * RAD;
    const float qscale = 4096.0f / RAD;
    const float* ptsB = points + (size_t)b * NF * 3;

    // pass 1: histogram of quantized in-radius distances
    for (int i = tid; i < NF; i += 256) {
        __builtin_prefetch(ptsB + (i + 1024) * 3, 0, 0);   // global_prefetch_b8
        float dx = ptsB[i*3+0] - cx;
        float dy = ptsB[i*3+1] - cy;
        float dz = ptsB[i*3+2] - cz;
        float d2 = dx*dx + dy*dy + dz*dz;
        if (d2 < r2) {
            float d = sqrtf(d2);
            int q = (int)(d * qscale);
            q = q > 4095 ? 4095 : q;
            atomicAdd(&hist[q], 1u);
        }
    }
    __syncthreads();

    // per-thread partial over 16 bins, then serial exclusive scan
    {
        unsigned int s = 0;
        for (int j = 0; j < 16; ++j) s += hist[tid*16 + j];
        scanp[tid] = s;
    }
    __syncthreads();
    if (tid == 0) {
        unsigned int run = 0;
        for (int t = 0; t < 256; ++t) { unsigned int v = scanp[t]; scanp[t] = run; run += v; }
    }
    __syncthreads();
    // find threshold bin T: first bin with cumulative >= P
    {
        unsigned int cum = scanp[tid];
        for (int j = 0; j < 16; ++j) {
            unsigned int h = hist[tid*16 + j];
            if (cum < PP && cum + h >= PP) atomicMin(&s_T, (unsigned int)(tid*16 + j));
            cum += h;
        }
    }
    __syncthreads();
    const unsigned int T = s_T;

    // pass 2: collect candidates with q <= T
    for (int i = tid; i < NF; i += 256) {
        __builtin_prefetch(ptsB + (i + 1024) * 3, 0, 0);
        float dx = ptsB[i*3+0] - cx;
        float dy = ptsB[i*3+1] - cy;
        float dz = ptsB[i*3+2] - cz;
        float d2 = dx*dx + dy*dy + dz*dz;
        if (d2 < r2) {
            float d = sqrtf(d2);
            int q = (int)(d * qscale);
            q = q > 4095 ? 4095 : q;
            if ((unsigned int)q <= T) {
                unsigned int pos = atomicAdd(&candCnt, 1u);
                if (pos < CAND_CAP) {
                    candKey[pos] = __float_as_uint(d);   // positive float: int-order == float-order
                    candIdx[pos] = (unsigned int)i;
                }
            }
        }
    }
    __syncthreads();
    const int ncand = (int)(candCnt < CAND_CAP ? candCnt : CAND_CAP);

    // exact rank-select among candidates (distance asc, tie by index)
    for (int t = tid; t < ncand; t += 256) {
        unsigned int myK = candKey[t], myI = candIdx[t];
        int rank = 0;
        for (int j = 0; j < ncand; ++j) {
            unsigned int kj = candKey[j];
            rank += (int)((kj < myK) | ((kj == myK) & (candIdx[j] < myI)));
        }
        if (rank < PP) selIdx[rank] = (int)myI;
    }
    __syncthreads();

    // gather features (zero-padded), points, mask
    const float* ftB = feats + (size_t)b * NF * DD;
    float* fo = ftOut + (size_t)bc * PP * DD;
    for (int e = tid; e < PP * DD; e += 256) {
        int p = e >> 6, dch = e & 63;
        int idx = selIdx[p];
        fo[e] = (idx >= 0) ? ftB[(size_t)idx * DD + dch] : 0.0f;
    }
    float* po = ptOut + (size_t)bc * PP * 3;
    for (int e = tid; e < PP * 3; e += 256) {
        int p = e / 3, k = e % 3;
        int idx = selIdx[p];
        po[e] = (idx >= 0) ? ptsB[(size_t)idx * 3 + k] : 0.0f;
    }
    if (tid < PP) maskOut[(size_t)bc * PP + tid] = (selIdx[tid] >= 0) ? 1 : 0;
}

// =====================================================================
// Kernel 2: normalize features + 64x64x64 cost GEMM via WMMA f16->f32.
// One block (128 thr = 4 waves) per (b,c); wave w owns M-tile rows
// [16w,16w+16).  WMMA results staged in LDS, then one coalesced,
// branch-free linear pass emits the masked augmented (65x65) matrix.
// =====================================================================
__global__ __launch_bounds__(128)
void cost_kernel(const float* __restrict__ sft,   // (B,NC,P,D)
                 const float* __restrict__ tft,
                 const int*   __restrict__ smask, // (B,NC,P)
                 const int*   __restrict__ tmask,
                 const float* __restrict__ alpha, // scalar
                 float* __restrict__ aug)         // (B,NC,65,65)
{
    __shared__ __align__(16) _Float16 sA[64 * 64];
    __shared__ __align__(16) _Float16 sB[64 * 64];
    __shared__ float costT[AUGN][AUGN + 1];   // only [0..63][0..63] written
    __shared__ int sm[AUGN], tm[AUGN];        // slot 64 forced to 0

    const int bc  = blockIdx.x;
    const int tid = threadIdx.x;

    // --- normalize: threads 0-63 -> src rows, 64-127 -> tgt rows ---
    {
        const int row = tid & 63;
        const float* src = (tid < 64) ? sft : tft;
        _Float16* dst = (tid < 64) ? sA : sB;
        const size_t base = ((size_t)bc * PP + row) * DD;
        float s = 0.0f;
        #pragma unroll 8
        for (int d0 = 0; d0 < DD; ++d0) { float v = src[base + d0]; s += v * v; }
        float inv = 1.0f / fmaxf(sqrtf(s), 1e-12f);
        #pragma unroll 8
        for (int d0 = 0; d0 < DD; ++d0) dst[row * 64 + d0] = (_Float16)(src[base + d0] * inv);
        if (tid < 64) sm[tid] = smask[(size_t)bc * PP + tid];
        else          tm[tid - 64] = tmask[(size_t)bc * PP + (tid - 64)];
        if (tid == 0) { sm[64] = 0; tm[64] = 0; }
    }
    __syncthreads();

    const int lane = tid & 31;
    const int wave = tid >> 5;
    const int m0   = wave * 16;
    const int rsel = lane & 15;
    const int kb   = (lane >> 4) << 3;       // 0 or 8

    #pragma unroll
    for (int n0 = 0; n0 < 64; n0 += 16) {
        v8f c = {};
        #pragma unroll
        for (int kk = 0; kk < 64; kk += 32) {
            // A fragment: lane holds row m0+rsel, halves 0-7 = K[kk+kb .. +7],
            // halves 8-15 = K[kk+kb+16 .. +23]   (per ISA 16-bit A layout)
            const _Float16* ap = &sA[(m0 + rsel) * 64 + kk + kb];
            const _Float16* bp = &sB[(n0 + rsel) * 64 + kk + kb];
            v8h alo = *(const v8h*)(ap);
            v8h ahi = *(const v8h*)(ap + 16);
            v8h blo = *(const v8h*)(bp);
            v8h bhi = *(const v8h*)(bp + 16);
            v16h a = __builtin_shufflevector(alo, ahi, 0,1,2,3,4,5,6,7,8,9,10,11,12,13,14,15);
            v16h bmat = __builtin_shufflevector(blo, bhi, 0,1,2,3,4,5,6,7,8,9,10,11,12,13,14,15);
            c = __builtin_amdgcn_wmma_f32_16x16x32_f16(
                    false, a, false, bmat, (short)0, c, false, false);
        }
        // C/D layout: VGPR r, lane<16 -> M=m0+r, lane>=16 -> M=m0+8+r; N=n0+(lane&15)
        const int pbase = m0 + ((lane < 16) ? 0 : 8);
        const int q     = n0 + rsel;
        #pragma unroll
        for (int r = 0; r < 8; ++r) costT[pbase + r][q] = c[r];
    }
    __syncthreads();

    // --- branch-free coalesced emission of augmented matrix ---
    const float alphav = alpha[0];
    float* augp = aug + (size_t)bc * AUGN * AUGN;
    for (int i = tid; i < AUGN * AUGN; i += 128) {
        int p = i / AUGN, q = i % AUGN;
        int pm = sm[p] & tm[q];                        // 0/1, unconditional LDS loads
        int dust = (p == 64) | (q == 64);
        float cval = costT[p][q];                      // garbage in dustbin lane, never selected
        float val = dust ? alphav : (pm ? cval : NEGV);
        augp[i] = val;
    }
}

// =====================================================================
// Kernel 3: Sinkhorn, 100 iterations, fully LDS-resident 65x66 tile.
// One block per (b,c); 65 active lanes (row pass / col pass).
// =====================================================================
__global__ __launch_bounds__(128)
void sinkhorn_kernel(const float* __restrict__ aug,  // (B,NC,65,65)
                     float* __restrict__ zOut)       // (B,NC,65,65) region of d_out
{
    __shared__ float M[AUGN][AUGN + 1];
    const int bc  = blockIdx.x;
    const int tid = threadIdx.x;
    const float* ap = aug + (size_t)bc * AUGN * AUGN;

    for (int i = tid; i < AUGN * AUGN; i += 128) M[i / AUGN][i % AUGN] = ap[i];
    __syncthreads();

    for (int it = 0; it < ITERS; ++it) {
        if (tid < AUGN) {                       // row logsumexp
            const int r = tid;
            float mx = -INFINITY;
            for (int q = 0; q < AUGN; ++q) mx = fmaxf(mx, M[r][q]);
            float s = 0.0f;
            for (int q = 0; q < AUGN; ++q) s += __expf(M[r][q] - mx);
            float lse = mx + __logf(s);
            for (int q = 0; q < AUGN; ++q) M[r][q] -= lse;
        }
        __syncthreads();
        if (tid < AUGN) {                       // col logsumexp
            const int cq = tid;
            float mx = -INFINITY;
            for (int r = 0; r < AUGN; ++r) mx = fmaxf(mx, M[r][cq]);
            float s = 0.0f;
            for (int r = 0; r < AUGN; ++r) s += __expf(M[r][cq] - mx);
            float lse = mx + __logf(s);
            for (int r = 0; r < AUGN; ++r) M[r][cq] -= lse;
        }
        __syncthreads();
    }

    float* zp = zOut + (size_t)bc * AUGN * AUGN;
    for (int i = tid; i < AUGN * AUGN; i += 128) zp[i] = __expf(M[i / AUGN][i % AUGN]);
}

// =====================================================================
// Kernel 4: mutual top-3 matching + correspondence emission.
// =====================================================================
__global__ __launch_bounds__(128)
void match_kernel(const float* __restrict__ zOut,   // (B,NC,65,65)
                  const int*   __restrict__ smask,
                  const int*   __restrict__ tmask,
                  const float* __restrict__ spt,    // (B,NC,P,3)
                  const float* __restrict__ tpt,
                  float* __restrict__ corrSrc,      // (B,NC*P*K,3)
                  float* __restrict__ corrTgt,
                  float* __restrict__ wf)           // (B,NC*P*K)
{
    __shared__ float Zm[PP][PP + 1];
    __shared__ float rowV[PP][KTOP];
    __shared__ int   rowI[PP][KTOP];
    __shared__ int   colI[PP][KTOP];
    __shared__ int   sm[PP], tm[PP];
    __shared__ int   s_valid;

    const int bc  = blockIdx.x;
    const int tid = threadIdx.x;
    const float* zp = zOut + (size_t)bc * AUGN * AUGN;

    if (tid < PP) sm[tid] = smask[(size_t)bc * PP + tid];
    else          tm[tid - PP] = tmask[(size_t)bc * PP + (tid - PP)];
    __syncthreads();

    for (int i = tid; i < PP * PP; i += 128) {
        int p = i >> 6, q = i & 63;
        float z = zp[p * AUGN + q];
        int pm = sm[p] & tm[q];                 // 0/1, no short-circuit branch
        Zm[p][q] = pm ? z : -1.0f;
    }
    if (tid == 0) {
        int va = 0, vb = 0;
        for (int j = 0; j < PP; ++j) { va |= sm[j]; vb |= tm[j]; }
        s_valid = (int)((va != 0) & (vb != 0));
    }
    __syncthreads();

    if (tid < PP) {               // row top-3 (value + index)
        const int r = tid;
        int i0 = 0, i1 = 0, i2 = 0; float b0 = -INFINITY, b1 = -INFINITY, b2 = -INFINITY;
        for (int q = 0; q < PP; ++q) { float v = Zm[r][q]; if (v > b0) { b0 = v; i0 = q; } }
        for (int q = 0; q < PP; ++q) { if (q == i0) continue; float v = Zm[r][q]; if (v > b1) { b1 = v; i1 = q; } }
        for (int q = 0; q < PP; ++q) { if (q == i0 || q == i1) continue; float v = Zm[r][q]; if (v > b2) { b2 = v; i2 = q; } }
        rowV[r][0] = b0; rowV[r][1] = b1; rowV[r][2] = b2;
        rowI[r][0] = i0; rowI[r][1] = i1; rowI[r][2] = i2;
    } else {                      // col top-3 (index only)
        const int cq = tid - PP;
        int i0 = 0, i1 = 0, i2 = 0; float b0 = -INFINITY, b1 = -INFINITY, b2 = -INFINITY;
        for (int r = 0; r < PP; ++r) { float v = Zm[r][cq]; if (v > b0) { b0 = v; i0 = r; } }
        for (int r = 0; r < PP; ++r) { if (r == i0) continue; float v = Zm[r][cq]; if (v > b1) { b1 = v; i1 = r; } }
        for (int r = 0; r < PP; ++r) { if (r == i0 || r == i1) continue; float v = Zm[r][cq]; if (v > b2) { b2 = v; i2 = r; } }
        colI[cq][0] = i0; colI[cq][1] = i1; colI[cq][2] = i2;
    }
    __syncthreads();

    const int valid = s_valid;
    for (int j = tid; j < PP * KTOP; j += 128) {
        int p  = j / KTOP;
        int ki = j % KTOP;
        int tgt = rowI[p][ki];
        float score = rowV[p][ki];
        int mut = (int)((colI[tgt][0] == p) | (colI[tgt][1] == p) | (colI[tgt][2] == p));
        float wfv = (mut & (int)(score > THC) & valid) ? 1.0f : 0.0f;
        size_t cix = (size_t)bc * (PP * KTOP) + j;     // == b*NC*P*K + c*P*K + j
        wf[cix] = wfv;
        const float* sp_ = spt + ((size_t)bc * PP + p) * 3;
        const float* tp_ = tpt + ((size_t)bc * PP + tgt) * 3;
        corrSrc[cix * 3 + 0] = sp_[0] * wfv;
        corrSrc[cix * 3 + 1] = sp_[1] * wfv;
        corrSrc[cix * 3 + 2] = sp_[2] * wfv;
        corrTgt[cix * 3 + 0] = tp_[0] * wfv;
        corrTgt[cix * 3 + 1] = tp_[1] * wfv;
        corrTgt[cix * 3 + 2] = tp_[2] * wfv;
    }
}

// =====================================================================
// Host launcher
// =====================================================================
extern "C" void kernel_launch(void* const* d_in, const int* in_sizes, int n_in,
                              void* d_out, int out_size, void* d_ws, size_t ws_size,
                              hipStream_t stream) {
    const float* src_ft  = (const float*)d_in[0];
    const float* tgt_ft  = (const float*)d_in[1];
    const float* src_pt  = (const float*)d_in[2];
    const float* tgt_pt  = (const float*)d_in[3];
    const int*   src_spi = (const int*)  d_in[4];
    const int*   tgt_spi = (const int*)  d_in[5];
    const float* src_cp  = (const float*)d_in[6];
    const float* tgt_cp  = (const float*)d_in[7];
    const float* alpha   = (const float*)d_in[8];

    // workspace layout (bytes)
    constexpr size_t SFT_B   = (size_t)BB * NC * PP * DD * 4;    // 8 MB
    constexpr size_t SPT_B   = (size_t)BB * NC * PP * 3  * 4;
    constexpr size_t MSK_B   = (size_t)BB * NC * PP * 4;
    constexpr size_t AUG_B   = (size_t)BB * NC * AUGN * AUGN * 4;
    char* ws = (char*)d_ws;
    float* w_sft   = (float*)(ws);
    float* w_tft   = (float*)(ws + SFT_B);
    float* w_spt   = (float*)(ws + 2*SFT_B);
    float* w_tpt   = (float*)(ws + 2*SFT_B + SPT_B);
    int*   w_smask = (int*)  (ws + 2*SFT_B + 2*SPT_B);
    int*   w_tmask = (int*)  (ws + 2*SFT_B + 2*SPT_B + MSK_B);
    float* w_aug   = (float*)(ws + 2*SFT_B + 2*SPT_B + 2*MSK_B);
    (void)AUG_B; (void)ws_size; (void)n_in; (void)in_sizes; (void)out_size;

    // output layout (floats): corr_src | corr_tgt | wf | Z
    constexpr size_t NCPK = (size_t)NC * PP * KTOP;              // 49152
    float* o_corr_src = (float*)d_out;
    float* o_corr_tgt = o_corr_src + (size_t)BB * NCPK * 3;
    float* o_wf       = o_corr_tgt + (size_t)BB * NCPK * 3;
    float* o_z        = o_wf       + (size_t)BB * NCPK;

    const int NBLK = BB * NC;   // 512

    gather_kernel<<<NBLK, 256, 0, stream>>>(src_ft, src_pt, src_spi, src_cp,
                                            w_sft, w_spt, w_smask);
    gather_kernel<<<NBLK, 256, 0, stream>>>(tgt_ft, tgt_pt, tgt_spi, tgt_cp,
                                            w_tft, w_tpt, w_tmask);
    cost_kernel<<<NBLK, 128, 0, stream>>>(w_sft, w_tft, w_smask, w_tmask, alpha, w_aug);
    sinkhorn_kernel<<<NBLK, 128, 0, stream>>>(w_aug, o_z);
    match_kernel<<<NBLK, 128, 0, stream>>>(o_z, w_smask, w_tmask, w_spt, w_tpt,
                                           o_corr_src, o_corr_tgt, o_wf);
}
